// LSTMmodel_10325101380324
// MI455X (gfx1250) — matrix operand order repaired and checked by reading, not compiled, not run on hardware
//
#include <hip/hip_runtime.h>

// Problem constants (from reference)
#define LB_   4
#define LWIN_ 128
#define CODE_ 64
#define H_    256
#define TENC_ (LB_*LWIN_)   // 512 encoder steps

typedef __attribute__((ext_vector_type(16))) __bf16 v16bf;
typedef __attribute__((ext_vector_type(8)))  float  v8f;
typedef __attribute__((ext_vector_type(8)))  int    v8i;
typedef __attribute__((ext_vector_type(4)))  int    v4i;

// Global-addrspace pointer so laundered addresses still emit global_load_b128
// (FLAT loads would tick both LOADcnt and DScnt and contend with LDS traffic).
typedef const __attribute__((address_space(1))) v4i* gbl_v4p;

// ---------------- packed workspace layout (units: u16 elements) ----------------
// B-operand layout per (ct, kb): [lane 0..31][e 0..15] bf16, where
//   N = ct*16 + (lane&15),  K = kb*32 + (lane>>4)*8 + (e&7) + (e>>3)*16
#define N_W0IH (64*2*512)            // Wih_0: N=1024 (64 ct), K=64 (2 kb)
#define N_HHP  (64*8*512)            // 1024 x 256 matrices
#define N_FCW  (4*8*512)             // fcW: N=64 (4 ct), K=256 (8 kb)
#define P_W0IH 0
#define P_W0HH (P_W0IH + N_W0IH)
#define P_W1IH (P_W0HH + N_HHP)
#define P_W1HH (P_W1IH + N_HHP)
#define P_FCW  (P_W1HH + N_HHP)
#define P_END  (P_FCW + N_FCW)       // 868352 u16 = 1736704 bytes (4B aligned)
// fp32 region (indices into ws viewed as float*)
#define F_BIAS0 (P_END/2)            // bih_0 + bhh_0   [1024]
#define F_BIAS1 (F_BIAS0 + 1024)     // bih_1 + bhh_1   [1024]
#define F_FCB   (F_BIAS1 + 1024)     // fcb             [64]
#define PACK_TOTAL (P_END + 2112)

// float -> bf16 bits, round-to-nearest-even
__device__ __forceinline__ unsigned short f2bf(float f) {
  unsigned int u = __builtin_bit_cast(unsigned int, f);
  u += 0x7FFFu + ((u >> 16) & 1u);
  return (unsigned short)(u >> 16);
}

__device__ __forceinline__ float sigmf(float x) { return 1.0f / (1.0f + __expf(-x)); }
__device__ __forceinline__ float tanh_f(float x) {
  float a = fabsf(x);
  float e = __expf(-2.0f * a);
  float t = (1.0f - e) / (1.0f + e);
  return x < 0.0f ? -t : t;
}

// ---------------- weight pre-pack kernel ----------------
__device__ __forceinline__ void pack_region(unsigned short* dst, const float* W,
                                            int K, int nkb, int idx) {
  int e   = idx & 15;
  int L   = (idx >> 4) & 31;
  int ckb = idx >> 9;               // ct*nkb + kb
  int kb  = ckb % nkb;
  int ct  = ckb / nkb;
  int n   = ct * 16 + (L & 15);
  int k   = kb * 32 + (L >> 4) * 8 + (e & 7) + (e >> 3) * 16;
  dst[idx] = f2bf(W[n * K + k]);
}

__global__ __launch_bounds__(256) void pack_kernel(
    const float* __restrict__ Wih0, const float* __restrict__ Whh0,
    const float* __restrict__ bih0, const float* __restrict__ bhh0,
    const float* __restrict__ Wih1, const float* __restrict__ Whh1,
    const float* __restrict__ bih1, const float* __restrict__ bhh1,
    const float* __restrict__ fcW,  const float* __restrict__ fcb,
    unsigned short* __restrict__ wsp, float* __restrict__ wsf)
{
  int idx = blockIdx.x * 256 + threadIdx.x;
  if (idx >= PACK_TOTAL) return;
  if (idx < P_W0HH)      pack_region(wsp + P_W0IH, Wih0, CODE_, 2, idx - P_W0IH);
  else if (idx < P_W1IH) pack_region(wsp + P_W0HH, Whh0, H_,    8, idx - P_W0HH);
  else if (idx < P_W1HH) pack_region(wsp + P_W1IH, Wih1, H_,    8, idx - P_W1IH);
  else if (idx < P_FCW)  pack_region(wsp + P_W1HH, Whh1, H_,    8, idx - P_W1HH);
  else if (idx < P_END)  pack_region(wsp + P_FCW,  fcW,  H_,    8, idx - P_FCW);
  else {
    int j = idx - P_END;
    if (j < 1024)      wsf[F_BIAS0 + j]          = bih0[j] + bhh0[j];
    else if (j < 2048) wsf[F_BIAS1 + (j - 1024)] = bih1[j - 1024] + bhh1[j - 1024];
    else               wsf[F_FCB   + (j - 2048)] = fcb[j - 2048];
  }
}

// ---------------- WMMA fragment loaders ----------------
__device__ __forceinline__ v16bf frag_from(gbl_v4p g) {
  v4i lo = g[0];
  v4i hi = g[1];
  v8i r = { lo[0], lo[1], lo[2], lo[3], hi[0], hi[1], hi[2], hi[3] };
  return __builtin_bit_cast(v16bf, r);
}

// Load all 8 B fragments of one kb-block. ONE laundered (opaque) base per block
// blocks LICM from hoisting these t-invariant loads out of the time loop (which
// previously caused scratch spills); the 8 q-tiles sit at compile-time constant
// byte offsets (q * NKB * 8192 < 2^23) that fold into the load IOFFSET.
template <int NKB>
__device__ __forceinline__ void load_bfrags(v16bf b[8], const unsigned short* p) {
  unsigned long a = (unsigned long)p;
  asm volatile("" : "+v"(a));
  #pragma unroll
  for (int q = 0; q < 8; ++q)
    b[q] = frag_from((gbl_v4p)(a + (unsigned long)(q * NKB * 8192)));
}

// Single B fragment (decoder FC), same laundering + global addrspace.
__device__ __forceinline__ v16bf load_bfrag(const unsigned short* p) {
  unsigned long a = (unsigned long)p;
  asm volatile("" : "+v"(a));
  return frag_from((gbl_v4p)a);
}

// A operand from LDS row-major bf16 [16][ld]:
// lane: M = lane&15, half = lane>>4; VGPR v holds K pair = k0 + (v>=4?16:0) + half*8 + 2*(v&3)
__device__ __forceinline__ v16bf load_afrag(const unsigned short* sh, int ld,
                                            int k0, int lane) {
  const int m    = lane & 15;
  const int half = lane >> 4;
  const unsigned short* row = sh + m * ld + k0 + half * 8;
  v8i r;
  #pragma unroll
  for (int v = 0; v < 8; ++v) {
    const int kk = ((v & 4) << 2) + 2 * (v & 3);
    r[v] = *(const int*)(row + kk);
  }
  return __builtin_bit_cast(v16bf, r);
}

// Accumulate acc[q] += A(shA) * B(Wp) for this wave's 8 column tiles.
// Per kb-block: 1 A-fragment (LDS), all 8 B loads issued before the 8 WMMAs so
// the backend overlaps L2 latency with prior WMMAs via partial loadcnt waits.
template <int NKB>
__device__ __forceinline__ void mm_accum(v8f acc[8], const unsigned short* shA, int ld,
                                         const unsigned short* __restrict__ Wp,
                                         int w, int lane) {
  #pragma unroll
  for (int kb = 0; kb < NKB; ++kb) {
    v16bf a = load_afrag(shA, ld, kb * 32, lane);
    v16bf b[8];
    load_bfrags<NKB>(b, Wp + ((w * NKB + kb) * 32 + lane) * 16);
    #pragma unroll
    for (int q = 0; q < 8; ++q)
      acc[q] = __builtin_amdgcn_wmma_f32_16x16x32_bf16(
          false, a, false, b[q], (short)0, acc[q], false, false);
  }
}

// ---------------- persistent recurrent kernel ----------------
// Grid: 16 WGs x 256 threads. WG owns batch rows [16*blockIdx.x, +16) for all steps.
// Wave w owns gate-column tiles ct = w + 8q (q=0..7) -> complete i/f/g/o quadruples
// for hidden units j in [16w,16w+16) U [16(w+8),16(w+8)+16).
__global__ __launch_bounds__(256, 1) void lstm_seq_kernel(
    const float* __restrict__ x, const float* __restrict__ tg,
    const unsigned short* __restrict__ wsp, const float* __restrict__ wsf,
    float* __restrict__ out)
{
  __shared__ __align__(16) unsigned short hA0[16 * H_];    // layer-0 h (bf16)
  __shared__ __align__(16) unsigned short hA1[16 * H_];    // layer-1 h (bf16)
  __shared__ __align__(16) unsigned short xA [16 * CODE_]; // step input (bf16)

  const int  tid  = threadIdx.x;
  const int  lane = tid & 31;
  const int  w    = tid >> 5;
  const int  n16  = lane & 15;
  const int  half = lane >> 4;
  const long b0   = (long)blockIdx.x * 16;

  for (int i = tid; i < 16 * H_; i += 256) { hA0[i] = 0; hA1[i] = 0; }

  float bias0[8], bias1[8];
  #pragma unroll
  for (int q = 0; q < 8; ++q) {
    const int col = 16 * (w + 8 * q) + n16;
    bias0[q] = wsf[F_BIAS0 + col];
    bias1[q] = wsf[F_BIAS1 + col];
  }
  const float fbias = (w < 4) ? wsf[F_FCB + 16 * w + n16] : 0.0f;

  float c0[2][8], c1[2][8];
  #pragma unroll
  for (int p = 0; p < 2; ++p)
    #pragma unroll
    for (int r = 0; r < 8; ++r) { c0[p][r] = 0.0f; c1[p][r] = 0.0f; }

  __syncthreads();

  for (int phase = 0; phase < 2; ++phase) {
    const int T = phase ? LWIN_ : TENC_;
    for (int t = 0; t < T; ++t) {
      // ---- stage layer-0 input tile (16 rows x 64) into xA as bf16 ----
      {
        const int m  = tid >> 4;
        const int cb = (tid & 15) * 4;
        const float* src;
        if (phase == 0)   src = x  + ((b0 + m) * TENC_ + t) * CODE_ + cb;
        else if (t == 0)  src = x  + ((b0 + m) * TENC_ + (TENC_ - 1)) * CODE_ + cb;
        else              src = tg + ((b0 + m) * LWIN_ + (t - 1)) * CODE_ + cb;
        float4 v = *(const float4*)src;
        unsigned long long pk =
            (unsigned long long)f2bf(v.x)
          | ((unsigned long long)f2bf(v.y) << 16)
          | ((unsigned long long)f2bf(v.z) << 32)
          | ((unsigned long long)f2bf(v.w) << 48);
        *(unsigned long long*)(xA + m * CODE_ + cb) = pk;
      }
      __syncthreads();

      // ---- layer 0 gates: x @ Wih0^T + h0 @ Whh0^T + b0 ----
      v8f acc[8];
      #pragma unroll
      for (int q = 0; q < 8; ++q) {
        v8f a;
        #pragma unroll
        for (int r = 0; r < 8; ++r) a[r] = bias0[q];
        acc[q] = a;
      }
      mm_accum<2>(acc, xA,  CODE_, wsp + P_W0IH, w, lane);
      mm_accum<8>(acc, hA0, H_,    wsp + P_W0HH, w, lane);
      __syncthreads();                 // all reads of hA0 done before overwrite
      #pragma unroll
      for (int p = 0; p < 2; ++p) {
        const int j = 16 * (w + 8 * p) + n16;
        #pragma unroll
        for (int r = 0; r < 8; ++r) {
          const float iv = sigmf(acc[0 + p][r]);
          const float fv = sigmf(acc[2 + p][r]);
          const float gv = tanh_f(acc[4 + p][r]);
          const float ov = sigmf(acc[6 + p][r]);
          const float cn = fv * c0[p][r] + iv * gv;
          c0[p][r] = cn;
          hA0[(r + 8 * half) * H_ + j] = f2bf(ov * tanh_f(cn));
        }
      }
      __syncthreads();

      // ---- layer 1 gates: h0 @ Wih1^T + h1 @ Whh1^T + b1 ----
      #pragma unroll
      for (int q = 0; q < 8; ++q) {
        v8f a;
        #pragma unroll
        for (int r = 0; r < 8; ++r) a[r] = bias1[q];
        acc[q] = a;
      }
      mm_accum<8>(acc, hA0, H_, wsp + P_W1IH, w, lane);
      mm_accum<8>(acc, hA1, H_, wsp + P_W1HH, w, lane);
      __syncthreads();                 // all reads of hA1 done before overwrite
      #pragma unroll
      for (int p = 0; p < 2; ++p) {
        const int j = 16 * (w + 8 * p) + n16;
        #pragma unroll
        for (int r = 0; r < 8; ++r) {
          const float iv = sigmf(acc[0 + p][r]);
          const float fv = sigmf(acc[2 + p][r]);
          const float gv = tanh_f(acc[4 + p][r]);
          const float ov = sigmf(acc[6 + p][r]);
          const float cn = fv * c1[p][r] + iv * gv;
          c1[p][r] = cn;
          hA1[(r + 8 * half) * H_ + j] = f2bf(ov * tanh_f(cn));
        }
      }
      __syncthreads();

      // ---- decoder output FC: out[b,t,:] = h1 @ fcW^T + fcb (waves 0..3) ----
      if (phase == 1 && w < 4) {       // wave-uniform: EXEC all-ones inside
        v8f fa;
        #pragma unroll
        for (int r = 0; r < 8; ++r) fa[r] = fbias;
        #pragma unroll
        for (int kb = 0; kb < 8; ++kb) {
          v16bf a = load_afrag(hA1, H_, kb * 32, lane);
          v16bf b = load_bfrag(wsp + P_FCW + ((w * 8 + kb) * 32 + lane) * 16);
          fa = __builtin_amdgcn_wmma_f32_16x16x32_bf16(
              false, a, false, b, (short)0, fa, false, false);
        }
        const int col = 16 * w + n16;
        #pragma unroll
        for (int r = 0; r < 8; ++r)
          out[(b0 + r + 8 * half) * (LWIN_ * CODE_) + (long)t * CODE_ + col] = fa[r];
      }
    }
  }
}

// ---------------- host entry ----------------
extern "C" void kernel_launch(void* const* d_in, const int* in_sizes, int n_in,
                              void* d_out, int out_size, void* d_ws, size_t ws_size,
                              hipStream_t stream) {
  const float* x    = (const float*)d_in[0];
  const float* tg   = (const float*)d_in[1];
  const float* Wih0 = (const float*)d_in[2];
  const float* Whh0 = (const float*)d_in[3];
  const float* bih0 = (const float*)d_in[4];
  const float* bhh0 = (const float*)d_in[5];
  const float* Wih1 = (const float*)d_in[6];
  const float* Whh1 = (const float*)d_in[7];
  const float* bih1 = (const float*)d_in[8];
  const float* bhh1 = (const float*)d_in[9];
  const float* fcW  = (const float*)d_in[10];
  const float* fcb  = (const float*)d_in[11];

  unsigned short* wsp = (unsigned short*)d_ws;
  float*          wsf = (float*)d_ws;

  pack_kernel<<<dim3((PACK_TOTAL + 255) / 256), dim3(256), 0, stream>>>(
      Wih0, Whh0, bih0, bhh0, Wih1, Whh1, bih1, bhh1, fcW, fcb, wsp, wsf);

  lstm_seq_kernel<<<dim3(16), dim3(256), 0, stream>>>(
      x, tg, wsp, wsf, (float*)d_out);
}
